// gemGAT_75986561401264
// MI455X (gfx1250) — compile-verified, matrix operand order checked by reading, not run on hardware
//
#include <hip/hip_runtime.h>
#include <hip/hip_bf16.h>

// Problem constants from the reference
#define NI_C 6000
#define NG_C 8000
#define NA_C 12000
#define H_C  8
#define D_C  64

typedef __attribute__((ext_vector_type(16))) __bf16 bf16x16;
typedef __attribute__((ext_vector_type(8)))  float  f32x8;

// ---------- small device helpers ----------
__device__ __forceinline__ unsigned fomap(float f) {
  unsigned u = __float_as_uint(f);
  return (u & 0x80000000u) ? ~u : (u | 0x80000000u);
}
__device__ __forceinline__ float founmap(unsigned m) {
  unsigned u = (m & 0x80000000u) ? (m & 0x7FFFFFFFu) : ~m;
  return __uint_as_float(u);
}
__device__ __forceinline__ float act_apply(float x, int act) {
  if (act == 1) return x > 0.f ? x : (__expf(x) - 1.f);   // ELU
  if (act == 2) return 1.f / (1.f + __expf(-x));          // sigmoid
  return x;
}
__device__ __forceinline__ bf16x16 pack16(float4 r0, float4 r1, float4 r2, float4 r3) {
  bf16x16 v;
  v[0]=(__bf16)r0.x;  v[1]=(__bf16)r0.y;  v[2]=(__bf16)r0.z;  v[3]=(__bf16)r0.w;
  v[4]=(__bf16)r1.x;  v[5]=(__bf16)r1.y;  v[6]=(__bf16)r1.z;  v[7]=(__bf16)r1.w;
  v[8]=(__bf16)r2.x;  v[9]=(__bf16)r2.y;  v[10]=(__bf16)r2.z; v[11]=(__bf16)r2.w;
  v[12]=(__bf16)r3.x; v[13]=(__bf16)r3.y; v[14]=(__bf16)r3.z; v[15]=(__bf16)r3.w;
  return v;
}

// ---------- WMMA GEMM: C[M x N] = act(A[M x K] * B[N x K]^T + bias[N]) ----------
// One wave computes a 32(M) x 16*NSUB(N) tile: 2 A-fragments, each B fragment is
// reused by two WMMAs. Block = 128 threads (4 waves stacked along M => 128 rows).
// Requires K%32==0, M%32==0; column tiling handled by NSUB + n_base.
// NSUB is a compile-time constant so all acc[] indices are static (no v_movrel).
template<int NSUB>
__global__ void wmma_gemm_k(const float* __restrict__ A, const float* __restrict__ B,
                            const float* __restrict__ bias, float* __restrict__ C,
                            int M, int N, int K, int act, int n_base)
{
  int lane = threadIdx.x & 31;
  int wave = threadIdx.x >> 5;
  int tm = blockIdx.y * 4 + wave;            // 32-row tile index
  if (tm * 32 >= M) return;                  // uniform per wave: EXEC stays all-ones
  int n0 = n_base + blockIdx.x * (16 * NSUB);
  int m = lane & 15, half = lane >> 4;

  const float* Arow0 = A + (size_t)(tm * 32 + m) * K;
  const float* Arow1 = Arow0 + (size_t)16 * K;
  f32x8 acc0[NSUB] = {};
  f32x8 acc1[NSUB] = {};

  for (int k0 = 0; k0 < K; k0 += 32) {
    // A fragment: lane m holds K = k0+8*half+[0..7] (v0..3) and +16+[0..7] (v4..7)
    const float4* pa0 = (const float4*)(Arow0 + k0 + half * 8);
    const float4* pa1 = (const float4*)(Arow1 + k0 + half * 8);
    bf16x16 af0 = pack16(pa0[0], pa0[1], pa0[4], pa0[5]);
    bf16x16 af1 = pack16(pa1[0], pa1[1], pa1[4], pa1[5]);
#pragma unroll
    for (int s = 0; s < NSUB; ++s) {
      const float* Brow = B + (size_t)(n0 + s * 16 + m) * K;   // column n of B^T
      const float4* pb = (const float4*)(Brow + k0 + half * 8);
      bf16x16 bf = pack16(pb[0], pb[1], pb[4], pb[5]);
      acc0[s] = __builtin_amdgcn_wmma_f32_16x16x32_bf16(
          false, af0, false, bf, (short)0, acc0[s], false, false);
      acc1[s] = __builtin_amdgcn_wmma_f32_16x16x32_bf16(
          false, af1, false, bf, (short)0, acc1[s], false, false);
    }
  }
  // C/D layout: row = v + 8*half, col = lane&15 within each 16x16 tile
#pragma unroll
  for (int s = 0; s < NSUB; ++s) {
    int col = n0 + s * 16 + (lane & 15);
    float bs = bias ? bias[col] : 0.f;
#pragma unroll
    for (int v = 0; v < 8; ++v) {
      int row = tm * 32 + v + 8 * half;
      C[(size_t)row * N + col] = act_apply(acc0[s][v] + bs, act);
    }
#pragma unroll
    for (int v = 0; v < 8; ++v) {
      int row = tm * 32 + 16 + v + 8 * half;
      C[(size_t)row * N + col] = act_apply(acc1[s][v] + bs, act);
    }
  }
}

// ---------- scalar fallback GEMM (tiny layers: K<32 or N<16) ----------
__global__ void simple_gemm_k(const float* __restrict__ A, const float* __restrict__ B,
                              const float* __restrict__ bias, float* __restrict__ C,
                              int M, int N, int K, int act)
{
  long t = (long)blockIdx.x * blockDim.x + threadIdx.x;
  if (t >= (long)M * N) return;
  int row = (int)(t / N), col = (int)(t % N);
  float s = bias ? bias[col] : 0.f;
  const float* a = A + (long)row * K;
  const float* b = B + (long)col * K;
  for (int k = 0; k < K; ++k) s += a[k] * b[k];
  C[t] = act_apply(s, act);
}

// ---------- GAT helper kernels ----------
// h[n][c] = x[n] * W[c]   (in_f == 1 projection; W is (C,1) flattened)
__global__ void outer1_k(const float* __restrict__ x, const float* __restrict__ W,
                         float* __restrict__ h, int N, int C)
{
  long t = (long)blockIdx.x * blockDim.x + threadIdx.x;
  if (t >= (long)N * C) return;
  int n = (int)(t / C), c = (int)(t % C);
  h[t] = x[n] * W[c];
}

// el[n][hh] = sum_o h[n,hh,o]*al[hh,o] ; er likewise
__global__ void elr_k(const float* __restrict__ h, const float* __restrict__ al,
                      const float* __restrict__ ar, float* __restrict__ el,
                      float* __restrict__ er, int N, int heads, int out)
{
  long t = (long)blockIdx.x * blockDim.x + threadIdx.x;
  if (t >= (long)N * heads) return;
  int hh = (int)(t % heads);
  long n = t / heads;
  const float* hp = h + (n * heads + hh) * out;
  const float* lp = al + (long)hh * out;
  const float* rp = ar + (long)hh * out;
  float sl = 0.f, sr = 0.f;
  for (int o = 0; o < out; ++o) { float v = hp[o]; sl += v * lp[o]; sr += v * rp[o]; }
  el[t] = sl; er[t] = sr;
}

__global__ void fill_f32_k(float* p, float v, long n) {
  long t = (long)blockIdx.x * blockDim.x + threadIdx.x;
  if (t < n) p[t] = v;
}
__global__ void fill_u32_k(unsigned* p, unsigned v, long n) {
  long t = (long)blockIdx.x * blockDim.x + threadIdx.x;
  if (t < n) p[t] = v;
}

__global__ void edge_max_k(const int* __restrict__ src, const int* __restrict__ dst,
                           const float* __restrict__ el, const float* __restrict__ er,
                           unsigned* __restrict__ mx, long EH, int heads)
{
  long t = (long)blockIdx.x * blockDim.x + threadIdx.x;
  if (t >= EH) return;
  long e = t / heads; int hh = (int)(t % heads);
  int s = src[e], d = dst[e];
  float v = el[(long)s * heads + hh] + er[(long)d * heads + hh];
  v = v > 0.f ? v : 0.2f * v;                       // leaky_relu(0.2)
  atomicMax(&mx[(long)d * heads + hh], fomap(v));
}

__global__ void edge_expsum_k(const int* __restrict__ src, const int* __restrict__ dst,
                              const float* __restrict__ el, const float* __restrict__ er,
                              const unsigned* __restrict__ mx, float* __restrict__ den,
                              float* __restrict__ ee, long EH, int heads)
{
  long t = (long)blockIdx.x * blockDim.x + threadIdx.x;
  if (t >= EH) return;
  long e = t / heads; int hh = (int)(t % heads);
  int s = src[e], d = dst[e];
  float v = el[(long)s * heads + hh] + er[(long)d * heads + hh];
  v = v > 0.f ? v : 0.2f * v;
  float mm = founmap(mx[(long)d * heads + hh]);
  float ev = __expf(v - mm);
  ee[t] = ev;
  atomicAdd(&den[(long)d * heads + hh], ev);
}

// rst[dst,hh,:] += (ee/den[dst,hh]) * h[src,hh,:]  (4 feats per thread)
__global__ void edge_agg_k(const int* __restrict__ src, const int* __restrict__ dst,
                           const float* __restrict__ ee, const float* __restrict__ den,
                           const float* __restrict__ h, float* __restrict__ rst,
                           long total, int heads, int out)
{
  long t = (long)blockIdx.x * blockDim.x + threadIdx.x;
  if (t >= total) return;
  int G = out >> 2;
  int g = (int)(t % G);
  long r = t / G;
  int hh = (int)(r % heads);
  long e = r / heads;
  int s = src[e], d = dst[e];
  float coef = ee[e * heads + hh] / den[(long)d * heads + hh];
  const float4 hv = *(const float4*)(h + ((long)s * heads + hh) * out + g * 4);
  float* o = rst + ((long)d * heads + hh) * out + g * 4;
  atomicAdd(o + 0, coef * hv.x);
  atomicAdd(o + 1, coef * hv.y);
  atomicAdd(o + 2, coef * hv.z);
  atomicAdd(o + 3, coef * hv.w);
}

__global__ void bias_act_k(float* __restrict__ y, const float* __restrict__ b,
                           long n, int C, int act)
{
  long t = (long)blockIdx.x * blockDim.x + threadIdx.x;
  if (t >= n) return;
  y[t] = act_apply(y[t] + b[(int)(t % C)], act);
}

__global__ void copy_f32_k(float* __restrict__ d, const float* __restrict__ s, long n) {
  long t = (long)blockIdx.x * blockDim.x + threadIdx.x;
  if (t < n) d[t] = s[t];
}

// g_all[n] = n < NI ? pred[n] : X[n]
__global__ void build_gall_k(float* __restrict__ g, const float* __restrict__ pred,
                             const float* __restrict__ X, int NI, int NA)
{
  int t = blockIdx.x * blockDim.x + threadIdx.x;
  if (t >= NA) return;
  g[t] = (t < NI) ? pred[t] : X[t];
}

// ---------- host orchestration ----------
extern "C" void kernel_launch(void* const* d_in, const int* in_sizes, int n_in,
                              void* d_out, int out_size, void* d_ws, size_t ws_size,
                              hipStream_t stream)
{
  (void)out_size; (void)ws_size;
  const int NI = NI_C, NG = NG_C, NA = NA_C, Hh = H_C, Dd = D_C;

  const float* X = (const float*)d_in[0];
  int cur = 1;
  auto F = [&]() { return (const float*)d_in[cur++]; };

  struct GatP { const float *W, *al, *ar, *b; };
  auto getGat = [&]() { GatP p; p.W = F(); p.al = F(); p.ar = F(); p.b = F(); return p; };
  GatP att1 = getGat(), out1 = getGat(), att2 = getGat(), out2 = getGat();
  GatP att3 = getGat(), out3 = getGat(), att4 = getGat(), out4 = getGat();
  GatP lp1 = getGat(), olp1 = getGat(), lp2 = getGat(), olp2 = getGat();
  GatP semi1 = getGat(), osemi1 = getGat(), semi2 = getGat(), osemi2 = getGat();

  struct Lin { const float *W, *b; int K, N; };
  const int mdims[9] = {64, 512, 256, 256, 256, 64, 16, 4, 1};
  const int ldims[4] = {64, 128, 128, 128};
  Lin pin[8], plink[3], pout[8];
  for (int i = 0; i < 8; ++i) { pin[i].W = F(); pin[i].b = F(); pin[i].K = mdims[i]; pin[i].N = mdims[i + 1]; }
  for (int i = 0; i < 3; ++i) { plink[i].W = F(); plink[i].b = F(); plink[i].K = ldims[i]; plink[i].N = ldims[i + 1]; }
  for (int i = 0; i < 8; ++i) { pout[i].W = F(); pout[i].b = F(); pout[i].K = mdims[i]; pout[i].N = mdims[i + 1]; }

  const int* g1s = (const int*)d_in[n_in - 6];
  const int* g1d = (const int*)d_in[n_in - 5];
  const int* g2s = (const int*)d_in[n_in - 4];
  const int* g2d = (const int*)d_in[n_in - 3];
  const int* g3s = (const int*)d_in[n_in - 2];
  const int* g3d = (const int*)d_in[n_in - 1];
  const int E1 = in_sizes[n_in - 6];
  const int E3 = in_sizes[n_in - 2];

  // workspace carve (256B aligned slices)
  char* w = (char*)d_ws;
  auto carve = [&](size_t bytes) { void* p = (void*)w; w += (bytes + 255) & ~(size_t)255; return p; };
  float*    bufA = (float*)carve((size_t)NA * 512 * 4);
  float*    bufB = (float*)carve((size_t)NA * 512 * 4);
  float*    el   = (float*)carve((size_t)NA * Hh * 4);
  float*    er   = (float*)carve((size_t)NA * Hh * 4);
  unsigned* mxb  = (unsigned*)carve((size_t)NA * Hh * 4);
  float*    den  = (float*)carve((size_t)NA * Hh * 4);
  float*    ee   = (float*)carve((size_t)E3 * Hh * 4);
  float*    gall = (float*)carve((size_t)NA * 4);

  auto B1 = [](long n, int bs) { return dim3((unsigned)((n + bs - 1) / bs)); };

  auto launch_gemm = [&](const float* A, const float* Bm, const float* bias, float* C,
                         int M, int N, int K, int act) {
    if ((K % 32 == 0) && (N % 16 == 0) && (M % 32 == 0)) {
      unsigned gy = (unsigned)((M + 127) / 128);       // 4 waves x 32 rows per block
      int n64 = N / 64;                                // full 4-subtile supertiles
      if (n64 > 0)
        wmma_gemm_k<4><<<dim3((unsigned)n64, gy), 128, 0, stream>>>(A, Bm, bias, C, M, N, K, act, 0);
      int rem = (N % 64) / 16;                         // trailing 16-col subtiles
      int nb = n64 * 64;
      if (rem == 1)
        wmma_gemm_k<1><<<dim3(1, gy), 128, 0, stream>>>(A, Bm, bias, C, M, N, K, act, nb);
      else if (rem == 2)
        wmma_gemm_k<2><<<dim3(1, gy), 128, 0, stream>>>(A, Bm, bias, C, M, N, K, act, nb);
      else if (rem == 3)
        wmma_gemm_k<3><<<dim3(1, gy), 128, 0, stream>>>(A, Bm, bias, C, M, N, K, act, nb);
    } else {
      simple_gemm_k<<<B1((long)M * N, 256), 256, 0, stream>>>(A, Bm, bias, C, M, N, K, act);
    }
  };

  auto run_gat = [&](const float* feat, int in_f, const GatP& p, const int* src, const int* dst,
                     int E, int N, int heads, int out, float* hbuf, float* obuf, int post_act) {
    int C = heads * out;
    if (in_f == 1)
      outer1_k<<<B1((long)N * C, 256), 256, 0, stream>>>(feat, p.W, hbuf, N, C);
    else
      launch_gemm(feat, p.W, nullptr, hbuf, N, C, in_f, 0);
    elr_k<<<B1((long)N * heads, 256), 256, 0, stream>>>(hbuf, p.al, p.ar, el, er, N, heads, out);
    fill_u32_k<<<B1((long)N * heads, 256), 256, 0, stream>>>(mxb, 0u, (long)N * heads);
    fill_f32_k<<<B1((long)N * heads, 256), 256, 0, stream>>>(den, 0.f, (long)N * heads);
    fill_f32_k<<<B1((long)N * C, 256), 256, 0, stream>>>(obuf, 0.f, (long)N * C);
    long EH = (long)E * heads;
    edge_max_k<<<B1(EH, 256), 256, 0, stream>>>(src, dst, el, er, mxb, EH, heads);
    edge_expsum_k<<<B1(EH, 256), 256, 0, stream>>>(src, dst, el, er, mxb, den, ee, EH, heads);
    long tot = EH * (out >> 2);
    edge_agg_k<<<B1(tot, 256), 256, 0, stream>>>(src, dst, ee, den, hbuf, obuf, tot, heads, out);
    bias_act_k<<<B1((long)N * C, 256), 256, 0, stream>>>(obuf, p.b, (long)N * C, C, post_act);
  };

  auto run_mlp = [&](const float* in, int rows, const Lin* Ls, int nL,
                     float* s0, float* s1) -> const float* {
    const float* curp = in;
    for (int i = 0; i < nL; ++i) {
      float* dst = (i & 1) ? s1 : s0;
      int act = (i < nL - 1) ? 1 : 0;   // ELU on all but last
      launch_gemm(curp, Ls[i].W, Ls[i].b, dst, rows, Ls[i].N, Ls[i].K, act);
      curp = dst;
    }
    return curp;
  };

  float* out_f = (float*)d_out;

  // ---- Phase 1: g1/g2 GAT stack on NG nodes ----
  run_gat(X,    1,   att1, g1s, g1d, E1, NG, Hh, Dd, bufB, bufA, 0);
  run_gat(bufA, 512, out1, g1s, g1d, E1, NG, 1,  Dd, bufB, bufA, 1);
  run_gat(bufA, 64,  att2, g1s, g1d, E1, NG, Hh, Dd, bufB, bufA, 0);
  run_gat(bufA, 512, out2, g1s, g1d, E1, NG, 1,  Dd, bufB, bufA, 1);
  run_gat(bufA, 64,  att3, g2s, g2d, E1, NG, Hh, Dd, bufB, bufA, 0);
  run_gat(bufA, 512, out3, g2s, g2d, E1, NG, 1,  Dd, bufB, bufA, 1);
  run_gat(bufA, 64,  att4, g2s, g2d, E1, NG, Hh, Dd, bufB, bufA, 0);
  run_gat(bufA, 512, out4, g2s, g2d, E1, NG, 1,  Dd, bufB, bufA, 1);

  // pred_in MLP on NG rows -> NG x 1 (final in bufA)
  const float* pred = run_mlp(bufA, NG, pin, 8, bufB, bufA);
  copy_f32_k<<<B1(NI, 256), 256, 0, stream>>>(out_f, pred, NI);          // g_in_pred
  build_gall_k<<<B1(NA, 256), 256, 0, stream>>>(gall, pred, X, NI, NA);  // g_all

  // ---- Phase 2: link-pred GAT stack on NA nodes ----
  run_gat(gall, 1,   lp1,  g3s, g3d, E3, NA, Hh, Dd, bufB, bufA, 0);
  run_gat(bufA, 512, olp1, g3s, g3d, E3, NA, 1,  Dd, bufB, bufA, 1);
  run_gat(bufA, 64,  lp2,  g3s, g3d, E3, NA, Hh, Dd, bufB, bufA, 0);
  run_gat(bufA, 512, olp2, g3s, g3d, E3, NA, 1,  Dd, bufB, bufA, 0);    // no ELU
  const float* zf = run_mlp(bufA, NA, plink, 3, bufB, bufA);            // NA x 128, ends in bufB
  const float* z1 = zf;
  const float* z2 = zf + (size_t)NG * 128;

  // ---- Dense sigmoid Gram matrices (bf16 WMMA, store-bound) ----
  float* o_s1  = out_f + NI + NA;                                // A_semi1:    8000 x 4000
  float* o_s2  = o_s1 + (size_t)NG * (NA - NG);                  // A_semi2:    4000 x 4000
  float* o_ori = o_s2 + (size_t)(NA - NG) * (NA - NG);           // A_semi_ori: 8000 x 8000
  launch_gemm(z1, z2, nullptr, o_s1, NG, NA - NG, 128, 2);
  launch_gemm(z2, z2, nullptr, o_s2, NA - NG, NA - NG, 128, 2);
  launch_gemm(z1, z1, nullptr, o_ori, NG, NG, 128, 2);

  // ---- Phase 3: semi GAT stack on NA nodes ----
  run_gat(gall, 1,   semi1,  g3s, g3d, E3, NA, Hh, Dd, bufB, bufA, 0);
  run_gat(bufA, 512, osemi1, g3s, g3d, E3, NA, 1,  Dd, bufB, bufA, 1);
  run_gat(bufA, 64,  semi2,  g3s, g3d, E3, NA, Hh, Dd, bufB, bufA, 0);
  run_gat(bufA, 512, osemi2, g3s, g3d, E3, NA, 1,  Dd, bufB, bufA, 0);  // no ELU
  const float* gp = run_mlp(bufA, NA, pout, 8, bufB, bufA);             // NA x 1
  copy_f32_k<<<B1(NA, 256), 256, 0, stream>>>(out_f + NI, gp, NA);      // g_pred_all
}
